// MultiHeadAttention_72619307041000
// MI455X (gfx1250) — matrix operand order repaired and checked by reading, not compiled, not run on hardware
//
#include <hip/hip_runtime.h>
#include <hip/hip_bf16.h>

// ---------------------------------------------------------------------------
// Multi-head attention forward for MI455X (gfx1250), bf16 WMMA path.
//   B=4, T=2048, C=1024, H=16, Dk=64.
// ~137 GFLOP vs ~0.3 GB of traffic at 23.3 TB/s -> compute bound, so all
// GEMMs go through v_wmma_f32_16x16x32_bf16 (f32 accumulation; bf16 keeps
// the fp32 exponent range for softmax inputs).
// Flash-attention kernel stages K/V tiles into LDS with CDNA5 async copies
// (global_load_async_to_lds_b128 / s_wait_asynccnt), double-buffered, since
// all 4 waves of a block share the same (b,h) K/V stream (4x traffic cut).
// ---------------------------------------------------------------------------

typedef __attribute__((ext_vector_type(16))) __bf16 v16bf;
typedef __attribute__((ext_vector_type(8)))  float  v8f;
typedef __attribute__((ext_vector_type(4)))  __bf16 v4bf;

#define DEV __device__ __forceinline__

constexpr int kB  = 4;
constexpr int kT  = 2048;
constexpr int kC  = 1024;
constexpr int kH  = 16;
constexpr int kDk = 64;
constexpr int kM  = kB * kT;              // 8192 rows
constexpr float kScale = 0.125f;          // 1/sqrt(Dk)

// ---------------- WMMA wrapper -------------------------------------------
DEV v8f wmma_bf16(v16bf a, v16bf b, v8f c) {
  // (neg_a, A, neg_b, B, c_mod, C, reuse_a, reuse_b)
  return __builtin_amdgcn_wmma_f32_16x16x32_bf16(false, a, false, b,
                                                 (short)0, c, false, false);
}

// ---------------- CDNA5 async copy: global -> LDS (ASYNCcnt) -------------
// VDST = per-lane LDS byte address (relative to workgroup LDS base),
// VADDR = per-lane 64-bit global address. One instruction moves 16B/lane.
DEV void async_copy_b128(uint32_t lds_off, uint64_t gaddr) {
  asm volatile("global_load_async_to_lds_b128 %0, %1, off"
               :: "v"(lds_off), "v"(gaddr)
               : "memory");
}
DEV void wait_async_le8() { asm volatile("s_wait_asynccnt 0x8" ::: "memory"); }
DEV void wait_async_0()   { asm volatile("s_wait_asynccnt 0x0" ::: "memory"); }

// ---------------- fragment loaders (per ISA VGPR layout tables) ----------
// A-matrix 16x32 bf16: lane L -> row = row0 + (L&15), half h = L>>4.
// VGPR 0..3 hold K = h*8 + {0..7}; VGPR 4..7 hold K = 16 + h*8 + {0..7}.
DEV v16bf load_a16x32(const __bf16* base, int ld, int row0, int k0) {
  const int lane = threadIdx.x & 31;
  const int m = row0 + (lane & 15);
  const int h = lane >> 4;
  const __bf16* p = base + (size_t)m * ld + k0 + h * 8;
  v16bf a;
  uint4* a4 = reinterpret_cast<uint4*>(&a);
  a4[0] = *reinterpret_cast<const uint4*>(p);
  a4[1] = *reinterpret_cast<const uint4*>(p + 16);
  return a;
}

// B-matrix 32x16 bf16: lane L -> col n = n0 + (L&15), half h = L>>4.
// VGPR 0..3 hold K = h*8 + {0..7}; VGPR 4..7 hold K = 16 + h*8 + {0..7}.
// Source laid out "K-pairs contiguous": element B[k][n] at base[n*ld + k].
DEV v16bf load_b32x16(const __bf16* base, int ld, int k0, int n0) {
  const int lane = threadIdx.x & 31;
  const int n = n0 + (lane & 15);
  const int h = lane >> 4;
  const __bf16* p = base + (size_t)n * ld + k0 + h * 8;
  v16bf b;
  uint4* b4 = reinterpret_cast<uint4*>(&b);
  b4[0] = *reinterpret_cast<const uint4*>(p);
  b4[1] = *reinterpret_cast<const uint4*>(p + 16);
  return b;
}

// ---------------- cross-lane row reductions (C-frag halves) --------------
DEV float half_reduce_max(float v) {
  v = fmaxf(v, __shfl_xor(v, 1, 32));
  v = fmaxf(v, __shfl_xor(v, 2, 32));
  v = fmaxf(v, __shfl_xor(v, 4, 32));
  v = fmaxf(v, __shfl_xor(v, 8, 32));
  return v;
}
DEV float half_reduce_add(float v) {
  v += __shfl_xor(v, 1, 32);
  v += __shfl_xor(v, 2, 32);
  v += __shfl_xor(v, 4, 32);
  v += __shfl_xor(v, 8, 32);
  return v;
}

// ---------------- fp32 -> bf16 cast kernel -------------------------------
__global__ __launch_bounds__(256) void cast_f32_bf16_kernel(
    const float* __restrict__ in, __bf16* __restrict__ out, int n4) {
  const int i = blockIdx.x * blockDim.x + threadIdx.x;
  if (i < n4) {
    const float4 f = reinterpret_cast<const float4*>(in)[i];
    v4bf o;
    o[0] = (__bf16)f.x; o[1] = (__bf16)f.y;
    o[2] = (__bf16)f.z; o[3] = (__bf16)f.w;
    reinterpret_cast<v4bf*>(out)[i] = o;
  }
}

// ---------------- projection GEMM: Y = X * W^T + bias --------------------
// X: [kM, kC] bf16 row-major. W: [kC(out d), kC(in c)] bf16 row-major, so
// B[k=c][n=d] = W[d*kC + c] -> K-pairs contiguous. 256 threads = 8 waves,
// each wave a 32x32 tile; block tile 128x64.
// MODE 0: bf16 out, [B,H,T,Dk]    (Q, K)
// MODE 1: bf16 out, [B,H,Dk,T]    (V transposed for the PV WMMA B-frag)
// MODE 2: f32  out, [kM, kC]      (final output)
template <int MODE>
__global__ __launch_bounds__(256) void proj_gemm_kernel(
    const __bf16* __restrict__ X, const __bf16* __restrict__ W,
    const float* __restrict__ bias, void* __restrict__ outv) {
  const int lane = threadIdx.x & 31;
  const int wid  = threadIdx.x >> 5;                 // 0..7
  const int row0 = blockIdx.y * 128 + (wid >> 1) * 32;
  const int col0 = blockIdx.x * 64  + (wid & 1) * 32;

  v8f acc[2][2] = {{{}, {}}, {{}, {}}};

  for (int k0 = 0; k0 < kC; k0 += 32) {
    if (k0 + 64 < kC) {  // pull next slabs toward L2 (global_prefetch_b8)
      __builtin_prefetch(X + (size_t)(row0 + (lane & 15)) * kC + k0 + 64, 0, 0);
      __builtin_prefetch(W + (size_t)(col0 + (lane & 15)) * kC + k0 + 64, 0, 0);
    }
    const v16bf a0 = load_a16x32(X, kC, row0,      k0);
    const v16bf a1 = load_a16x32(X, kC, row0 + 16, k0);
    const v16bf b0 = load_b32x16(W, kC, k0, col0);
    const v16bf b1 = load_b32x16(W, kC, k0, col0 + 16);
    acc[0][0] = wmma_bf16(a0, b0, acc[0][0]);
    acc[0][1] = wmma_bf16(a0, b1, acc[0][1]);
    acc[1][0] = wmma_bf16(a1, b0, acc[1][0]);
    acc[1][1] = wmma_bf16(a1, b1, acc[1][1]);
  }

  const int hh = lane >> 4;
  const int cl = lane & 15;
#pragma unroll
  for (int i = 0; i < 2; ++i) {
#pragma unroll
    for (int j = 0; j < 2; ++j) {
      const int n  = col0 + j * 16 + cl;
      const float bv = bias[n];
#pragma unroll
      for (int r = 0; r < 8; ++r) {
        const int m = row0 + i * 16 + r + 8 * hh;   // C-frag row layout
        const float y = acc[i][j][r] + bv;
        if (MODE == 2) {
          reinterpret_cast<float*>(outv)[(size_t)m * kC + n] = y;
        } else {
          const int b = m >> 11, t = m & (kT - 1);
          const int head = n >> 6, d = n & (kDk - 1);
          if (MODE == 0)
            reinterpret_cast<__bf16*>(outv)
                [((size_t)(b * kH + head) * kT + t) * kDk + d] = (__bf16)y;
          else
            reinterpret_cast<__bf16*>(outv)
                [((size_t)(b * kH + head) * kDk + d) * kT + t] = (__bf16)y;
        }
      }
    }
  }
}

// ---------------- flash attention ----------------------------------------
// Block = 4 waves = 64 query rows of one (b,h). K/V tiles (64 keys) are
// async-copied into double-buffered LDS once per block (all waves share
// them). Per 32-key slab: 4 WMMAs for S, online softmax, P re-layout
// through LDS (C-frag -> A-frag), 4 WMMAs of P against V^T.
//
// LDS map (single array, base offset 0, byte offsets):
//   buf b (b=0,1): K tile [64 keys][64 d]  at b*16384        (8192 B)
//                  V tile [64 d][64 keys]  at b*16384 + 8192 (8192 B)
//   P scratch: wave w at 32768 + w*1024                      (4*1024 B)
__global__ __launch_bounds__(128) void flash_attn_kernel(
    const __bf16* __restrict__ Q, const __bf16* __restrict__ K,
    const __bf16* __restrict__ Vt, __bf16* __restrict__ Ctx) {
  __shared__ __align__(16) __bf16 smem[18432];   // 36 KB

  const int tid  = threadIdx.x;
  const int lane = tid & 31;
  const int wid  = tid >> 5;
  const int bh   = blockIdx.y;                  // b*16 + h
  const int b    = bh >> 4;
  const int h    = bh & 15;
  const int row0 = blockIdx.x * 64 + wid * 16;

  const __bf16* Qb = Q  + (size_t)bh * kT * kDk;
  const __bf16* Kb = K  + (size_t)bh * kT * kDk;
  const __bf16* Vb = Vt + (size_t)bh * kDk * kT;
  const uint64_t KbBytes = (uint64_t)(uintptr_t)Kb;
  const uint64_t VbBytes = (uint64_t)(uintptr_t)Vb;

  // Stage one 64-key K/V tile into LDS buffer `buf` (8 async b128 / thread).
  auto stage = [&](int s0, int buf) {
    const uint32_t kl = (uint32_t)(buf * 16384);
    const uint64_t kg = KbBytes + (uint64_t)s0 * (kDk * 2);  // contiguous 8KB
#pragma unroll
    for (int c = 0; c < 4; ++c) {
      const int ch = tid + c * 128;              // 512 chunks of 16B
      async_copy_b128(kl + ch * 16, kg + (uint32_t)ch * 16);
    }
    const uint32_t vl = kl + 8192;
    const uint64_t vg = VbBytes + (uint64_t)s0 * 2;
#pragma unroll
    for (int c = 0; c < 4; ++c) {
      const int ch = tid + c * 128;
      const int d = ch >> 3, cc = ch & 7;        // Vt row d, 16B chunk cc
      async_copy_b128(vl + ch * 16,
                      vg + (uint64_t)d * (kT * 2) + (uint32_t)cc * 16);
    }
  };

  const v16bf qa0 = load_a16x32(Qb, kDk, row0, 0);
  const v16bf qa1 = load_a16x32(Qb, kDk, row0, 32);

  v8f o[4] = {{}, {}, {}, {}};
  float mi[8], li[8];
#pragma unroll
  for (int r = 0; r < 8; ++r) { mi[r] = -3.0e38f; li[r] = 0.0f; }

  __bf16* myP = smem + 16384 + wid * 512;
  const int hh = lane >> 4;
  const int cl = lane & 15;

  stage(0, 0);

  constexpr int kTiles = kT / 64;
  for (int it = 0; it < kTiles; ++it) {
    const int buf = it & 1;
    if (it + 1 < kTiles) {
      stage((it + 1) * 64, buf ^ 1);   // prefetch next tile into other buffer
      wait_async_le8();                // first 8 (current tile) have landed
    } else {
      wait_async_0();
    }
    __syncthreads();                   // publish tile to all waves

    const __bf16* Ktile = smem + buf * 8192;   // [64 keys][64 d]
    const __bf16* Vtile = Ktile + 4096;        // [64 d][64 keys]

#pragma unroll
    for (int ss = 0; ss < 64; ss += 32) {
      // --- S = Q * K^T for a 16x32 slab (two 16-col C tiles) ---
      v8f sa[2];
#pragma unroll
      for (int j = 0; j < 2; ++j) {
        const v16bf kb0 = load_b32x16(Ktile, kDk, 0,  ss + 16 * j);
        const v16bf kb1 = load_b32x16(Ktile, kDk, 32, ss + 16 * j);
        v8f acc = {};
        acc = wmma_bf16(qa0, kb0, acc);
        acc = wmma_bf16(qa1, kb1, acc);
        sa[j] = acc;
      }

      // --- online softmax update ---
      float corr[8];
#pragma unroll
      for (int r = 0; r < 8; ++r) {
        const float v0 = sa[0][r] * kScale;
        const float v1 = sa[1][r] * kScale;
        const float mx   = half_reduce_max(fmaxf(v0, v1));
        const float mnew = fmaxf(mi[r], mx);
        corr[r] = __expf(mi[r] - mnew);
        const float p0 = __expf(v0 - mnew);
        const float p1 = __expf(v1 - mnew);
        li[r] = li[r] * corr[r] + half_reduce_add(p0 + p1);
        mi[r] = mnew;
        sa[0][r] = p0;
        sa[1][r] = p1;
      }

      // rescale running O
#pragma unroll
      for (int dt = 0; dt < 4; ++dt)
#pragma unroll
        for (int r = 0; r < 8; ++r) o[dt][r] *= corr[r];

      // --- P: C-frag layout -> LDS (16x32 row-major bf16) -> A-frag ---
#pragma unroll
      for (int j = 0; j < 2; ++j)
#pragma unroll
        for (int r = 0; r < 8; ++r)
          myP[(r + 8 * hh) * 32 + j * 16 + cl] = (__bf16)sa[j][r];

      const v16bf pa = load_a16x32(myP, 32, 0, 0);

      // --- O += P * V (Vtile[d][s], K-pairs over s contiguous) ---
#pragma unroll
      for (int dt = 0; dt < 4; ++dt) {
        const v16bf vb = load_b32x16(Vtile, 64, ss, dt * 16);
        o[dt] = wmma_bf16(pa, vb, o[dt]);
      }
    }
    __syncthreads();                   // done with this buffer block-wide
  }

  // --- epilogue: normalize and write ctx [B, T, C] (heads re-merged) ---
#pragma unroll
  for (int r = 0; r < 8; ++r) {
    const float inv = 1.0f / li[r];
    const int t = row0 + r + 8 * hh;
#pragma unroll
    for (int dt = 0; dt < 4; ++dt) {
      Ctx[((size_t)(b * kT + t)) * kC + h * kDk + dt * 16 + cl] =
          (__bf16)(o[dt][r] * inv);
    }
  }
}

// ---------------- host launcher ------------------------------------------
extern "C" void kernel_launch(void* const* d_in, const int* in_sizes, int n_in,
                              void* d_out, int out_size, void* d_ws, size_t ws_size,
                              hipStream_t stream) {
  (void)in_sizes; (void)n_in; (void)out_size; (void)ws_size;

  const float* x  = (const float*)d_in[0];
  const float* Wq = (const float*)d_in[1];
  const float* bq = (const float*)d_in[2];
  const float* Wk = (const float*)d_in[3];
  const float* bk = (const float*)d_in[4];
  const float* Wv = (const float*)d_in[5];
  const float* bv = (const float*)d_in[6];
  const float* Wo = (const float*)d_in[7];
  const float* bo = (const float*)d_in[8];

  char* ws = (char*)d_ws;
  size_t off = 0;
  auto carve = [&](size_t bytes) -> void* {
    void* p = ws + off;
    off += (bytes + 255) & ~(size_t)255;
    return p;
  };
  __bf16* Xbf  = (__bf16*)carve((size_t)kM * kC * 2);   // 16 MB
  __bf16* Wqb  = (__bf16*)carve((size_t)kC * kC * 2);   //  2 MB
  __bf16* Wkb  = (__bf16*)carve((size_t)kC * kC * 2);
  __bf16* Wvb  = (__bf16*)carve((size_t)kC * kC * 2);
  __bf16* Wob  = (__bf16*)carve((size_t)kC * kC * 2);
  __bf16* Qbf  = (__bf16*)carve((size_t)kM * kC * 2);   // [B,H,T,Dk]
  __bf16* Kbf  = (__bf16*)carve((size_t)kM * kC * 2);   // [B,H,T,Dk]
  __bf16* Vtbf = (__bf16*)carve((size_t)kM * kC * 2);   // [B,H,Dk,T]
  __bf16* Ctxb = (__bf16*)carve((size_t)kM * kC * 2);   // [B,T,C]

  auto cast = [&](const float* in, __bf16* out, size_t n) {
    const int n4 = (int)(n / 4);
    cast_f32_bf16_kernel<<<(n4 + 255) / 256, 256, 0, stream>>>(in, out, n4);
  };
  cast(x,  Xbf, (size_t)kM * kC);
  cast(Wq, Wqb, (size_t)kC * kC);
  cast(Wk, Wkb, (size_t)kC * kC);
  cast(Wv, Wvb, (size_t)kC * kC);
  cast(Wo, Wob, (size_t)kC * kC);

  const dim3 pgrid(kC / 64, kM / 128);
  const dim3 pblk(256);
  proj_gemm_kernel<0><<<pgrid, pblk, 0, stream>>>(Xbf, Wqb, bq, Qbf);
  proj_gemm_kernel<0><<<pgrid, pblk, 0, stream>>>(Xbf, Wkb, bk, Kbf);
  proj_gemm_kernel<1><<<pgrid, pblk, 0, stream>>>(Xbf, Wvb, bv, Vtbf);

  flash_attn_kernel<<<dim3(kT / 64, kB * kH), 128, 0, stream>>>(Qbf, Kbf, Vtbf, Ctxb);

  proj_gemm_kernel<2><<<pgrid, pblk, 0, stream>>>(Ctxb, Wob, bo, d_out);
}
